// GNNLayer_37452114821373
// MI455X (gfx1250) — matrix operand order repaired and checked by reading, not compiled, not run on hardware
//
#include <hip/hip_runtime.h>
#include <hip/hip_bf16.h>

typedef __attribute__((ext_vector_type(16))) _Float16 v16h;
typedef __attribute__((ext_vector_type(8)))  float    v8f;

// ---------------------------------------------------------------------------
// Hardware tanh (gfx1250 V_TANH_F32 transcendental). Sigmoid via tanh.
// ---------------------------------------------------------------------------
__device__ __forceinline__ float htanh(float x) {
#if __has_builtin(__builtin_amdgcn_tanhf)
    return __builtin_amdgcn_tanhf(x);
#elif __has_builtin(__builtin_amdgcn_tanh_f32)
    return __builtin_amdgcn_tanh_f32(x);
#else
    float r;
    asm volatile("v_tanh_f32 %0, %1" : "=v"(r) : "v"(x));
    return r;
#endif
}
__device__ __forceinline__ float hsigmoid(float x) {
    return 0.5f * (1.0f + htanh(0.5f * x));
}

// ---------------------------------------------------------------------------
// Weight pack: fp32 (K x N, row-major) -> f16 WMMA *A-operand* fragments.
// Fragment index = nt*KT + kt ; within fragment: lane (0..31), half h (0..15).
// A operand as packed here (f16 16x32, contiguous k order): lane L -> M(out-neuron) = nt*16 + (L&15);
//   half h -> K = kt*32 + (L>>4)*16 + h.
// Out-of-range (K,N) -> 0 (pads N=5 -> 16 and N=1 -> 16).
// ---------------------------------------------------------------------------
__global__ void __launch_bounds__(256)
pack_w_kernel(const float* __restrict__ W, _Float16* __restrict__ out,
              int K, int N, int KT, int NT) {
    int idx   = blockIdx.x * 256 + threadIdx.x;
    int total = KT * NT * 512;
    if (idx >= total) return;
    int h    = idx & 15;
    int r    = h >> 1;
    int p    = h & 1;
    int lane = (idx >> 4) & 31;
    int frag = idx >> 9;
    int kt   = frag % KT;
    int nt   = frag / KT;
    int n = nt * 16 + (lane & 15);
    int k = kt * 32 + (lane >> 4) * 16 + h;
    float v = (k < K && n < N) ? W[(size_t)k * N + n] : 0.0f;
    out[idx] = (_Float16)v;
}

// ---------------------------------------------------------------------------
// Dense layer over a 64-row LDS tile (row stride 256 halves).
// Weights = A operand (KT frags resident in regs per N-tile);
// activations = B operand, one contiguous 32B LDS read per (kt, row-tile):
//   B layout: lane -> row = rt*16 + (L&15); halves -> K = kt*32 + (L>>4)*16 + h.
// D layout: lane -> row; element i -> neuron = nt*16 + i + 8*(L>>4); the 8
// outputs per lane are contiguous halves -> one 16B LDS store.
// ---------------------------------------------------------------------------
template <int KT>
__device__ __forceinline__ void
dense_layer_tanh(const _Float16* __restrict__ inT, _Float16* __restrict__ outT,
                 const _Float16* __restrict__ wpack,
                 const float* __restrict__ bias, int lane) {
    const int m16 = lane & 15;
    const int g   = lane >> 4;
    for (int nt = 0; nt < 16; ++nt) {
        const v16h* wv = reinterpret_cast<const v16h*>(wpack) +
                         (size_t)nt * KT * 32 + lane;
        v16h aw[KT];
#pragma unroll
        for (int kt = 0; kt < KT; ++kt) aw[kt] = wv[(size_t)kt * 32];
        float bv[8];
#pragma unroll
        for (int i = 0; i < 8; ++i) bv[i] = bias[nt * 16 + i + 8 * g];
#pragma unroll
        for (int rt = 0; rt < 4; ++rt) {
            const _Float16* bt = inT + (rt * 16 + m16) * 256;
            v8f c = {0.f, 0.f, 0.f, 0.f, 0.f, 0.f, 0.f, 0.f};
#pragma unroll
            for (int kt = 0; kt < KT; ++kt) {
                v16h bf = *reinterpret_cast<const v16h*>(bt + kt * 32 + g * 16);
                c = __builtin_amdgcn_wmma_f32_16x16x32_f16(
                    false, aw[kt], false, bf, (short)0, c, false, false);
            }
            union { _Float16 q[8]; uint4 u; } o;
#pragma unroll
            for (int i = 0; i < 8; ++i)
                o.q[i] = (_Float16)htanh(c[i] + bv[i]);
            *reinterpret_cast<uint4*>(
                outT + (rt * 16 + m16) * 256 + nt * 16 + 8 * g) = o.u;
        }
    }
}

// ---------------------------------------------------------------------------
// MLP1: edge_attr viewed as [N,256]; 256->256->256->256->5
// (tanh x3, sigmoid). One wave per block, 64 rows per wave, LDS double buffer.
// ---------------------------------------------------------------------------
__global__ void __launch_bounds__(32)
mlp1_kernel(const float* __restrict__ ein,
            const _Float16* __restrict__ w0, const float* __restrict__ b0,
            const _Float16* __restrict__ w1, const float* __restrict__ b1,
            const _Float16* __restrict__ w2, const float* __restrict__ b2,
            const _Float16* __restrict__ w3, const float* __restrict__ b3,
            float* __restrict__ out1, int nrows) {
    __shared__ alignas(32) _Float16 tiles[2][64 * 256];
    const int lane = threadIdx.x & 31;
    const int m16  = lane & 15;
    const int g    = lane >> 4;
    long row0 = (long)blockIdx.x * 64;
    if (row0 > nrows - 64) row0 = nrows - 64;  // uniform clamp, EXEC stays full
    _Float16* cur = tiles[0];
    _Float16* nxt = tiles[1];

    __builtin_prefetch(w0, 0, 0);

    // Stage 64x256 fp32 -> f16 into LDS (exact trip count, 8B stores).
    const float4* src = reinterpret_cast<const float4*>(ein + row0 * 256);
    for (int i = lane; i < 4096; i += 32) {
        float4 v = src[i];
        union { _Float16 q[4]; unsigned long long u; } o;
        o.q[0] = (_Float16)v.x; o.q[1] = (_Float16)v.y;
        o.q[2] = (_Float16)v.z; o.q[3] = (_Float16)v.w;
        *reinterpret_cast<unsigned long long*>(cur + i * 4) = o.u;
    }
    __syncthreads();

    dense_layer_tanh<8>(cur, nxt, w0, b0, lane); __syncthreads();
    dense_layer_tanh<8>(nxt, cur, w1, b1, lane); __syncthreads();
    dense_layer_tanh<8>(cur, nxt, w2, b2, lane); __syncthreads();

    // Final layer: 256 -> 5 (padded to 16), sigmoid; store real columns only.
    {
        const v16h* wv = reinterpret_cast<const v16h*>(w3) + lane;
        v16h aw[8];
#pragma unroll
        for (int kt = 0; kt < 8; ++kt) aw[kt] = wv[(size_t)kt * 32];
#pragma unroll
        for (int rt = 0; rt < 4; ++rt) {
            const _Float16* bt = nxt + (rt * 16 + m16) * 256;
            v8f c = {0.f, 0.f, 0.f, 0.f, 0.f, 0.f, 0.f, 0.f};
#pragma unroll
            for (int kt = 0; kt < 8; ++kt) {
                v16h bf = *reinterpret_cast<const v16h*>(bt + kt * 32 + g * 16);
                c = __builtin_amdgcn_wmma_f32_16x16x32_f16(
                    false, aw[kt], false, bf, (short)0, c, false, false);
            }
            if (g == 0) {  // neurons 0..7 live in g==0 lanes; real N = 5
                long row = row0 + rt * 16 + m16;
#pragma unroll
                for (int i = 0; i < 5; ++i)
                    out1[row * 5 + i] = hsigmoid(c[i] + b3[i]);
            }
        }
    }
}

// ---------------------------------------------------------------------------
// MLP2: gather out1 through edge_index[1] into [N,160] rows;
// 160->256->256->256->1 (tanh x3, sigmoid).
// ---------------------------------------------------------------------------
__global__ void __launch_bounds__(32)
mlp2_kernel(const float* __restrict__ out1,
            const int* __restrict__ tgt,  // edge_index row 1, length N*32
            const _Float16* __restrict__ w0, const float* __restrict__ b0,
            const _Float16* __restrict__ w1, const float* __restrict__ b1,
            const _Float16* __restrict__ w2, const float* __restrict__ b2,
            const _Float16* __restrict__ w3, const float* __restrict__ b3,
            float* __restrict__ out, int nrows) {
    __shared__ alignas(32) _Float16 tiles[2][64 * 256];
    const int lane = threadIdx.x & 31;
    const int m16  = lane & 15;
    const int g    = lane >> 4;
    long row0 = (long)blockIdx.x * 64;
    if (row0 > nrows - 64) row0 = nrows - 64;
    _Float16* cur = tiles[0];
    _Float16* nxt = tiles[1];

    __builtin_prefetch(w0, 0, 0);

    // Gather stage: 64 rows x 32 edges x 5 values -> 64x160 f16 (stride 256).
    for (int t = lane; t < 2048; t += 32) {
        int m = t >> 5, j = t & 31;
        long e  = (row0 + m) * 32 + j;
        int  nd = tgt[e];
        const float* s = out1 + (long)nd * 5;
        _Float16* d = cur + m * 256 + j * 5;
        d[0] = (_Float16)s[0]; d[1] = (_Float16)s[1]; d[2] = (_Float16)s[2];
        d[3] = (_Float16)s[3]; d[4] = (_Float16)s[4];
    }
    __syncthreads();

    dense_layer_tanh<5>(cur, nxt, w0, b0, lane); __syncthreads();
    dense_layer_tanh<8>(nxt, cur, w1, b1, lane); __syncthreads();
    dense_layer_tanh<8>(cur, nxt, w2, b2, lane); __syncthreads();

    // Final layer: 256 -> 1 (padded to 16), sigmoid, neuron 0 only.
    {
        const v16h* wv = reinterpret_cast<const v16h*>(w3) + lane;
        v16h aw[8];
#pragma unroll
        for (int kt = 0; kt < 8; ++kt) aw[kt] = wv[(size_t)kt * 32];
#pragma unroll
        for (int rt = 0; rt < 4; ++rt) {
            const _Float16* bt = nxt + (rt * 16 + m16) * 256;
            v8f c = {0.f, 0.f, 0.f, 0.f, 0.f, 0.f, 0.f, 0.f};
#pragma unroll
            for (int kt = 0; kt < 8; ++kt) {
                v16h bf = *reinterpret_cast<const v16h*>(bt + kt * 32 + g * 16);
                c = __builtin_amdgcn_wmma_f32_16x16x32_f16(
                    false, aw[kt], false, bf, (short)0, c, false, false);
            }
            if (g == 0)
                out[row0 + rt * 16 + m16] = hsigmoid(c[0] + b3[0]);
        }
    }
}

// ---------------------------------------------------------------------------
// Host launcher. Inputs (setup dict order):
// 0:x 1:edge_index 2:edge_attr 3..10: w1_0,b1_0..w1_3,b1_3 11..18: w2_*,b2_*
// ---------------------------------------------------------------------------
extern "C" void kernel_launch(void* const* d_in, const int* in_sizes, int n_in,
                              void* d_out, int out_size, void* d_ws,
                              size_t ws_size, hipStream_t stream) {
    const int*   edge_index = (const int*)d_in[1];
    const float* edge_attr  = (const float*)d_in[2];
    const float* w1w[4] = {(const float*)d_in[3], (const float*)d_in[5],
                           (const float*)d_in[7], (const float*)d_in[9]};
    const float* w1b[4] = {(const float*)d_in[4], (const float*)d_in[6],
                           (const float*)d_in[8], (const float*)d_in[10]};
    const float* w2w[4] = {(const float*)d_in[11], (const float*)d_in[13],
                           (const float*)d_in[15], (const float*)d_in[17]};
    const float* w2b[4] = {(const float*)d_in[12], (const float*)d_in[14],
                           (const float*)d_in[16], (const float*)d_in[18]};

    const int  N = in_sizes[0] / 8;        // 100000 nodes (x is [N,8])
    const long E = (long)in_sizes[1] / 2;  // edges (edge_index is [2,E])

    // Workspace layout (halves): packed f16 weights, then fp32 out1 buffer.
    _Float16* wp = (_Float16*)d_ws;
    const size_t FR = 512;  // halves per fragment
    size_t o10 = 0;
    size_t o11 = o10 + 8 * 16 * FR;
    size_t o12 = o11 + 8 * 16 * FR;
    size_t o13 = o12 + 8 * 16 * FR;
    size_t o20 = o13 + 8 * 1 * FR;
    size_t o21 = o20 + 5 * 16 * FR;
    size_t o22 = o21 + 8 * 16 * FR;
    size_t o23 = o22 + 8 * 16 * FR;
    size_t oend = o23 + 8 * 1 * FR;        // multiple of 512 halves
    float* out1buf = (float*)(wp + oend);  // [N,5] fp32

    auto pack = [&](const float* W, size_t off, int K, int Nn, int KT, int NT) {
        int total = KT * NT * 512;
        pack_w_kernel<<<(total + 255) / 256, 256, 0, stream>>>(
            W, wp + off, K, Nn, KT, NT);
    };
    pack(w1w[0], o10, 256, 256, 8, 16);
    pack(w1w[1], o11, 256, 256, 8, 16);
    pack(w1w[2], o12, 256, 256, 8, 16);
    pack(w1w[3], o13, 256,   5, 8,  1);
    pack(w2w[0], o20, 160, 256, 5, 16);
    pack(w2w[1], o21, 256, 256, 8, 16);
    pack(w2w[2], o22, 256, 256, 8, 16);
    pack(w2w[3], o23, 256,   1, 8,  1);

    const int nblocks = (N + 63) / 64;  // 1 wave/block, 64 rows/wave

    mlp1_kernel<<<nblocks, 32, 0, stream>>>(
        edge_attr,
        wp + o10, w1b[0], wp + o11, w1b[1],
        wp + o12, w1b[2], wp + o13, w1b[3],
        out1buf, N);

    mlp2_kernel<<<nblocks, 32, 0, stream>>>(
        out1buf, edge_index + E,
        wp + o20, w2b[0], wp + o21, w2b[1],
        wp + o22, w2b[2], wp + o23, w2b[3],
        (float*)d_out, N);
}
